// AnalysisModel_75892072121043
// MI455X (gfx1250) — matrix operand order repaired and checked by reading, not compiled, not run on hardware
//
#include <hip/hip_runtime.h>
#include <hip/hip_fp16.h>
#include <math.h>

#define PHI_F      1.618033988749895f
#define TWO_PI_F   6.283185307179586f
#define LUT_N      4096
#define LUT_STEP   (TWO_PI_F / 4096.0f)
#define LUT_SCALE  (4096.0f / TWO_PI_F)

typedef __attribute__((ext_vector_type(16))) _Float16 v16h;
typedef __attribute__((ext_vector_type(8)))  _Float16 v8h;
typedef __attribute__((ext_vector_type(8)))  float    v8f;

// 4096-entry sin/cos LUT with linear interpolation (table values generated on
// the fly from the grid definition -> identical math to the reference table).
__device__ __forceinline__ void lut_sincos(float th, float& s, float& c) {
    float pos = th * LUT_SCALE;
    float i0f = floorf(pos);
    float fr  = pos - i0f;
    int i0 = ((int)i0f) & (LUT_N - 1);
    int i1 = (i0 + 1) & (LUT_N - 1);
    float s0, c0, s1, c1;
    __sincosf((float)i0 * LUT_STEP, &s0, &c0);
    __sincosf((float)i1 * LUT_STEP, &s1, &c1);
    s = s0 + (s1 - s0) * fr;
    c = c0 + (c1 - c0) * fr;
}

// ---------------------------------------------------------------- scan ------
// One thread per (b, d); sequential over t.  states[b,t,d] = nr + ni.
__global__ void scan_kernel(const int* __restrict__ ids,
                            const float* __restrict__ emb,
                            float* __restrict__ states) {
    int b = blockIdx.x;        // 0..7
    int d = threadIdx.x;       // 0..63
    float hr = 0.f, hi = 0.f;
    for (int t = 0; t < 512; ++t) {
        int tok   = ids[b * 512 + t];
        float w   = emb[tok * 128 + d];
        float bb  = emb[tok * 128 + 64 + d];
        float wl  = 1.0f + fabsf(w);
        float tp  = (float)t * PHI_F;
        float thr = hr / wl + bb + tp;
        float thi = hi / wl + bb + tp;
        float sr, cr, si, ci;
        lut_sincos(thr, sr, cr);
        lut_sincos(thi, si, ci);
        float nr = cr * ci - sr * si;
        float ni = cr * si + sr * ci;
        hr = nr; hi = ni;
        states[(b * 512 + t) * 64 + d] = nr + ni;
    }
}

// ---------------------------------------------------------------- q/k -------
// q[b,t,h,:] = [cos(th_q), sin(th_q)] (2*dh=32), k likewise (no t*phi term).
__global__ void qk_kernel(const float* __restrict__ states,
                          const float* __restrict__ wq, const float* __restrict__ bq,
                          const float* __restrict__ wk, const float* __restrict__ bk,
                          float* __restrict__ q, float* __restrict__ k) {
    int i  = blockIdx.x * blockDim.x + threadIdx.x;   // < 8*512*4*16 = 262144
    int j  = i & 15;
    int h  = (i >> 4) & 3;
    int bt = i >> 6;
    int t  = bt & 511;
    int wi = h * 16 + j;
    float xp = states[bt * 64 + h * 16 + j];
    float thq = xp / (1.f + fabsf(wq[wi])) + bq[wi] + (float)t * PHI_F;
    float thk = xp / (1.f + fabsf(wk[wi])) + bk[wi];
    float sq, cq, sk, ck;
    lut_sincos(thq, sq, cq);
    lut_sincos(thk, sk, ck);
    int base = (bt * 4 + h) * 32;
    q[base + j] = cq;  q[base + 16 + j] = sq;
    k[base + j] = ck;  k[base + 16 + j] = sk;
}

// ----------------------------------------------------- block reduction ------
__device__ __forceinline__ float blk_reduce(float v, float* red, bool do_max) {
    #pragma unroll
    for (int o = 16; o > 0; o >>= 1) {
        float other = __shfl_down(v, o, 32);
        v = do_max ? fmaxf(v, other) : (v + other);
    }
    int wid  = threadIdx.x >> 5;
    int lane = threadIdx.x & 31;
    if (lane == 0) red[wid] = v;
    __syncthreads();
    float r;
    if (do_max)
        r = fmaxf(fmaxf(fmaxf(red[0], red[1]), fmaxf(red[2], red[3])),
                  fmaxf(fmaxf(red[4], red[5]), fmaxf(red[6], red[7])));
    else
        r = red[0] + red[1] + red[2] + red[3] + red[4] + red[5] + red[6] + red[7];
    __syncthreads();
    return r;
}

// ------------------------------------------------------------ attention -----
// One block per (b,t).  Head-summed softmax weights accumulated in LDS, then
// ctx[d] = sum_{s<t} wsum[s] * states[b,s,d].  Strict-causal (s < t).
__global__ __launch_bounds__(256)
void attn_kernel(const float* __restrict__ states,
                 const float* __restrict__ q, const float* __restrict__ k,
                 float* __restrict__ ctx) {
    __shared__ float wsum[512];
    __shared__ float red[8];
    int bt  = blockIdx.x;
    int b   = bt >> 9;
    int t   = bt & 511;
    int tid = threadIdx.x;
    if (t == 0) {                       // masked row -> has_ctx zeroes it
        if (tid < 64) ctx[bt * 64 + tid] = 0.f;
        return;
    }
    for (int i = tid; i < 512; i += 256) wsum[i] = 0.f;
    __syncthreads();
    const float inv_scale = 0.17677669529663688f;  // 1/sqrt(2*dh)
    for (int h = 0; h < 4; ++h) {
        const float* qv = q + (bt * 4 + h) * 32;
        float sc0 = -INFINITY, sc1 = -INFINITY;
        int s0 = tid, s1 = tid + 256;
        if (s0 < t) {
            const float* kv = k + ((b * 512 + s0) * 4 + h) * 32;
            float acc = 0.f;
            #pragma unroll
            for (int j = 0; j < 32; ++j) acc += qv[j] * kv[j];
            sc0 = acc * inv_scale;
        }
        if (s1 < t) {
            const float* kv = k + ((b * 512 + s1) * 4 + h) * 32;
            float acc = 0.f;
            #pragma unroll
            for (int j = 0; j < 32; ++j) acc += qv[j] * kv[j];
            sc1 = acc * inv_scale;
        }
        float m   = blk_reduce(fmaxf(sc0, sc1), red, true);
        float e0  = (s0 < t) ? __expf(sc0 - m) : 0.f;
        float e1  = (s1 < t) ? __expf(sc1 - m) : 0.f;
        float sum = blk_reduce(e0 + e1, red, false);
        float inv = 1.0f / sum;
        if (s0 < t) wsum[s0] += e0 * inv;   // slots are thread-exclusive
        if (s1 < t) wsum[s1] += e1 * inv;
        __syncthreads();
    }
    if (tid < 64) {
        float acc = 0.f;
        for (int s = 0; s < t; ++s) acc += wsum[s] * states[(b * 512 + s) * 64 + tid];
        ctx[bt * 64 + tid] = acc;
    }
}

// ------------------------------------------------------------- post ---------
// x = states + has_ctx * ctx @ w_ctx^T; resonator bank; SiLU; emit x as f16.
__global__ __launch_bounds__(128)
void post_kernel(const float* __restrict__ states, const float* __restrict__ ctx,
                 const float* __restrict__ w_ctx,  const float* __restrict__ W_res,
                 const float* __restrict__ B_res,  const float* __restrict__ w_pr,
                 const float* __restrict__ w_pi,   _Float16* __restrict__ xh) {
    __shared__ float ctxs[64];
    __shared__ float xv[64];
    __shared__ float cs[128];
    __shared__ float ss[128];
    int bt  = blockIdx.x;
    int t   = bt & 511;
    int tid = threadIdx.x;
    if (tid < 64) ctxs[tid] = ctx[bt * 64 + tid];
    __syncthreads();
    if (tid < 64) {
        float acc = 0.f;
        if (t > 0) {
            #pragma unroll 8
            for (int j = 0; j < 64; ++j) acc += ctxs[j] * w_ctx[tid * 64 + j];
        }
        xv[tid] = states[bt * 64 + tid] + acc;
    }
    __syncthreads();
    {   // resonator bank: one thread per n
        int n = tid;
        float tp = (float)t * PHI_F;
        float csum = 0.f, ssum = 0.f;
        for (int d = 0; d < 64; ++d) {
            float th = xv[d] / (1.f + fabsf(W_res[n * 64 + d])) + B_res[n * 64 + d] + tp;
            float s, c;
            lut_sincos(th, s, c);
            csum += c; ssum += s;
        }
        cs[n] = csum; ss[n] = ssum;
    }
    __syncthreads();
    if (tid < 64) {
        float g = 0.f;
        #pragma unroll 8
        for (int n = 0; n < 128; ++n)
            g += cs[n] * w_pr[tid * 128 + n] + ss[n] * w_pi[tid * 128 + n];
        float sil = g / (1.f + __expf(-g));
        xh[bt * 64 + tid] = (_Float16)(xv[tid] + sil);
    }
}

// ------------------------------------------------------- weight cast --------
__global__ void wconv_kernel(const float* __restrict__ w, _Float16* __restrict__ wh, int n) {
    int i = blockIdx.x * blockDim.x + threadIdx.x;
    if (i < n) wh[i] = (_Float16)w[i];
}

// --------------------------------------------------------------- GEMM -------
// out(4096 x 32000) = xh(4096 x 64) * wh(32000 x 64)^T  via WMMA f16.
// Block = 8 waves -> 32(M) x 128(N) tile; each wave: 16x32 via 4 WMMAs.
__global__ __launch_bounds__(256)
void gemm_kernel(const _Float16* __restrict__ xh, const _Float16* __restrict__ wh,
                 float* __restrict__ out) {
    const int V = 32000;
    int wave = threadIdx.x >> 5;
    int lane = threadIdx.x & 31;
    int m_base  = blockIdx.y * 32 + (wave & 1) * 16;
    int n_base  = blockIdx.x * 128 + (wave >> 1) * 32;
    int mlane = lane & 15;
    int kh    = lane >> 4;          // K half-group select

    // A fragment, contiguous k order (same as B): lanes 0-15 -> K 0..15,
    // lanes 16-31 -> K 16..31; two K-blocks (koff 0, 32) for K=64.
    const _Float16* arow = xh + (m_base + mlane) * 64;
    union { v16h v; v8h h[2]; } a0, a1, b00, b01, b10, b11;
    a0.h[0] = *(const v8h*)(arow + kh * 16);
    a0.h[1] = *(const v8h*)(arow + kh * 16 + 8);
    a1.h[0] = *(const v8h*)(arow + 32 + kh * 16);
    a1.h[1] = *(const v8h*)(arow + 32 + kh * 16 + 8);

    // B fragment, 32x16: lane n = lane&15, lanes 0-15 hold K 0..15,
    // lanes 16-31 hold K 16..31 (contiguous in w_out rows = K-major).
    const _Float16* brow0 = wh + (n_base + mlane) * 64;
    const _Float16* brow1 = wh + (n_base + 16 + mlane) * 64;
    b00.h[0] = *(const v8h*)(brow0 + kh * 16);
    b00.h[1] = *(const v8h*)(brow0 + kh * 16 + 8);
    b01.h[0] = *(const v8h*)(brow0 + 32 + kh * 16);
    b01.h[1] = *(const v8h*)(brow0 + 32 + kh * 16 + 8);
    b10.h[0] = *(const v8h*)(brow1 + kh * 16);
    b10.h[1] = *(const v8h*)(brow1 + kh * 16 + 8);
    b11.h[0] = *(const v8h*)(brow1 + 32 + kh * 16);
    b11.h[1] = *(const v8h*)(brow1 + 32 + kh * 16 + 8);

    v8f c0 = {}, c1 = {};
    c0 = __builtin_amdgcn_wmma_f32_16x16x32_f16(false, a0.v, false, b00.v, (short)0, c0, false, false);
    c0 = __builtin_amdgcn_wmma_f32_16x16x32_f16(false, a1.v, false, b01.v, (short)0, c0, false, false);
    c1 = __builtin_amdgcn_wmma_f32_16x16x32_f16(false, a0.v, false, b10.v, (short)0, c1, false, false);
    c1 = __builtin_amdgcn_wmma_f32_16x16x32_f16(false, a1.v, false, b11.v, (short)0, c1, false, false);

    // C/D layout: VGPR j -> M = j (+8 for lanes 16-31), N = lane&15.
    int row0 = m_base + (kh ? 8 : 0);
    #pragma unroll
    for (int j = 0; j < 8; ++j) {
        out[(row0 + j) * V + n_base + mlane]      = c0[j];
        out[(row0 + j) * V + n_base + 16 + mlane] = c1[j];
    }
}

// ------------------------------------------------------------- launch -------
extern "C" void kernel_launch(void* const* d_in, const int* in_sizes, int n_in,
                              void* d_out, int out_size, void* d_ws, size_t ws_size,
                              hipStream_t stream) {
    (void)in_sizes; (void)n_in; (void)out_size; (void)ws_size;
    const int*   ids   = (const int*)  d_in[0];
    const float* emb   = (const float*)d_in[1];
    const float* wq    = (const float*)d_in[2];
    const float* bq    = (const float*)d_in[3];
    const float* wk    = (const float*)d_in[4];
    const float* bk    = (const float*)d_in[5];
    const float* w_ctx = (const float*)d_in[6];
    const float* W_res = (const float*)d_in[7];
    const float* B_res = (const float*)d_in[8];
    const float* w_pr  = (const float*)d_in[9];
    const float* w_pi  = (const float*)d_in[10];
    const float* w_out = (const float*)d_in[11];
    float* out = (float*)d_out;

    // workspace layout (~11 MB)
    float* states = (float*)d_ws;            // 262144 f32
    float* qbuf   = states + 262144;         // 524288 f32
    float* kbuf   = qbuf + 524288;           // 524288 f32
    float* ctx    = kbuf + 524288;           // 262144 f32
    _Float16* xh  = (_Float16*)(ctx + 262144);   // 262144 f16
    _Float16* wh  = xh + 262144;                  // 2048000 f16

    scan_kernel <<<8, 64, 0, stream>>>(ids, emb, states);
    qk_kernel   <<<1024, 256, 0, stream>>>(states, wq, bq, wk, bk, qbuf, kbuf);
    wconv_kernel<<<8000, 256, 0, stream>>>(w_out, wh, 2048000);
    attn_kernel <<<4096, 256, 0, stream>>>(states, qbuf, kbuf, ctx);
    post_kernel <<<4096, 128, 0, stream>>>(states, ctx, w_ctx, W_res, B_res, w_pr, w_pi, xh);
    gemm_kernel <<<dim3(250, 128), 256, 0, stream>>>(xh, wh, out);
}
